// Encoder_Decoder_3075196584122
// MI455X (gfx1250) — compile-verified, matrix operand order checked by reading, not run on hardware
//
#include <hip/hip_runtime.h>
#include <cstdint>
#include <cstddef>

// ---------- types ----------
typedef __bf16 bf16_t;
typedef __attribute__((ext_vector_type(16))) bf16_t v16bf;
typedef __attribute__((ext_vector_type(8)))  bf16_t v8bf;
typedef __attribute__((ext_vector_type(8)))  float  v8f;

#define SEQ    256
#define BATCH  128
#define HID    1024
#define G4H    4096     // 4*HID
#define KT32   32       // K tiles of 32 along HID

// ---------- device helpers ----------
__device__ __forceinline__ float fast_sig(float x) {
    return 1.0f / (1.0f + __expf(-x));
}
__device__ __forceinline__ float fast_tanh(float x) {
    float ax = fabsf(x);
    float e  = __expf(-2.0f * ax);
    float r  = (1.0f - e) / (1.0f + e);
    return copysignf(r, x);
}

// A fragment: 16x32 bf16 tile, row-major source with leading dim ld (elements).
// ISA layout: lanes 0-15 -> M=lane, K = {k0..k0+7, k0+16..k0+23}
//             lanes 16-31 -> M=lane-16, K = {k0+8..k0+15, k0+24..k0+31}
__device__ __forceinline__ v16bf load_a_frag(const bf16_t* p, int ld) {
    const int lane = threadIdx.x & 31;
    const int r    = lane & 15;
    const int kh   = lane >> 4;               // 0 or 1
    const bf16_t* row = p + (size_t)r * ld + kh * 8;
    v8bf lo = *(const v8bf*)(row);            // K offsets +0..7 (or +8..15)
    v8bf hi = *(const v8bf*)(row + 16);       // K offsets +16..23 (or +24..31)
    return __builtin_shufflevector(lo, hi, 0,1,2,3,4,5,6,7,8,9,10,11,12,13,14,15);
}

// B fragment from pre-swizzled weight storage: fragIdx = nTile*KT32 + kTile
__device__ __forceinline__ v16bf load_b_frag(const bf16_t* frag, int fragIdx) {
    const int lane = threadIdx.x & 31;
    return *(const v16bf*)(frag + (size_t)fragIdx * 512 + (size_t)lane * 16);
}

__device__ __forceinline__ v8f wmma_bf16(v16bf a, v16bf b, v8f c) {
    return __builtin_amdgcn_wmma_f32_16x16x32_bf16(false, a, false, b,
                                                   (short)0, c, false, false);
}

// ---------- kernel 1: zero h0/c, fold biases ----------
__global__ __launch_bounds__(256) void init_state_kernel(
    float* c, bf16_t* h0, float* bias, const float* b_ih, const float* b_hh) {
    int i = blockIdx.x * 256 + threadIdx.x;       // 131072 threads exact
    c[i]  = 0.0f;
    h0[i] = (bf16_t)0.0f;
    if (i < G4H) bias[i] = b_ih[i] + b_hh[i];
}

// ---------- kernel 2: f32 weights [4096,1024] -> bf16 WMMA-B fragments ----------
// frag layout: [nTile(256)][kTile(32)][lane(32)][e(16)], element = W[n*1024+k]
__global__ __launch_bounds__(256) void convert_weight_kernel(
    const float* __restrict__ W, bf16_t* __restrict__ frag) {
    unsigned tid = blockIdx.x * 256 + threadIdx.x;  // 4,194,304 exact
    int e    = tid & 15;
    int lane = (tid >> 4) & 31;
    int kt   = (tid >> 9) & 31;
    int nt   = tid >> 14;
    int n  = nt * 16 + (lane & 15);
    int kh = lane >> 4;
    int k  = kt * 32 + kh * 8 + ((e < 8) ? e : 8 + e);
    frag[tid] = (bf16_t)W[(size_t)n * HID + k];
}

// ---------- kernel 3: embedding gather + f32->bf16  (X[t*B+b][k]) ----------
__global__ __launch_bounds__(256) void gather_embed_kernel(
    const int* __restrict__ idx, const float* __restrict__ embed,
    bf16_t* __restrict__ X) {
    size_t t4  = (size_t)(blockIdx.x * 256 + threadIdx.x) * 4;  // 32768 blocks exact
    size_t row = t4 >> 10;
    int    k   = (int)(t4 & 1023);
    int tok = idx[row];
    const float4 v = *(const float4*)(embed + (size_t)tok * HID + k);
    bf16_t* o = X + row * HID + k;
    o[0] = (bf16_t)v.x; o[1] = (bf16_t)v.y; o[2] = (bf16_t)v.z; o[3] = (bf16_t)v.w;
}

// ---------- kernel 4: Gin = X @ Wih^T + bias   (M=32768,N=4096,K=1024) ----------
// wave tile: 64(M) x 64(N); 32768 waves -> 4096 blocks x 8 waves.
// 16 f32 accumulators (128 VGPRs); A fragment is transient per m-subtile.
// Gin is streamed (written once, read once) -> non-temporal stores keep the
// 192MB L2 free for the weight fragments + X that the recurrent phase reuses.
__global__ __launch_bounds__(256) void input_gemm_kernel(
    const bf16_t* __restrict__ X, const bf16_t* __restrict__ Wfrag,
    const float* __restrict__ bias, float* __restrict__ Gin) {
    const int wid   = threadIdx.x >> 5;
    const int lane  = threadIdx.x & 31;
    const int wtile = blockIdx.x * 8 + wid;
    const int nG = wtile & 63;          // 64 groups of 64 cols
    const int mG = wtile >> 6;          // 512 groups of 64 rows
    const int m0 = mG * 64, n0 = nG * 64;
    const int col = lane & 15;
    const int rowHi = (lane >> 4) * 8;

    v8f acc[4][4];                      // [mi][ni]
    #pragma unroll
    for (int ni = 0; ni < 4; ++ni) {
        float bv = bias[n0 + ni * 16 + col];
        #pragma unroll
        for (int mi = 0; mi < 4; ++mi) {
            #pragma unroll
            for (int j = 0; j < 8; ++j) acc[mi][ni][j] = bv;
        }
    }

    for (int kt = 0; kt < KT32; ++kt) {
        v16bf b[4];
        #pragma unroll
        for (int ni = 0; ni < 4; ++ni)
            b[ni] = load_b_frag(Wfrag, ((n0 >> 4) + ni) * KT32 + kt);
        #pragma unroll
        for (int mi = 0; mi < 4; ++mi) {
            v16bf a = load_a_frag(X + (size_t)(m0 + mi * 16) * HID + kt * 32, HID);
            #pragma unroll
            for (int ni = 0; ni < 4; ++ni)
                acc[mi][ni] = wmma_bf16(a, b[ni], acc[mi][ni]);
        }
    }

    #pragma unroll
    for (int mi = 0; mi < 4; ++mi)
        #pragma unroll
        for (int ni = 0; ni < 4; ++ni)
            #pragma unroll
            for (int r = 0; r < 8; ++r) {
                int row = m0 + mi * 16 + rowHi + r;
                __builtin_nontemporal_store(
                    acc[mi][ni][r],
                    &Gin[(size_t)row * G4H + n0 + ni * 16 + col]);
            }
}

// ---------- kernel 5: one recurrent step, fully fused ----------
// gates = Gin[t] + h @ Whh^T ; LSTM pointwise ; writes h (f32 out + bf16 next)
// wave tile: 16(M) x 16(N) x 4 gates (N offsets 0,H,2H,3H) -> 512 waves.
// Gin loads and hs stores are non-temporal streams; Whh fragments, h and c
// stay L2/WGP$-resident across all 256 steps.
__global__ __launch_bounds__(128) void lstm_step_kernel(
    const float* __restrict__ Gin_t, const bf16_t* __restrict__ Whh_frag,
    const bf16_t* __restrict__ h_in, bf16_t* __restrict__ h_out,
    float* __restrict__ c, float* __restrict__ out_t) {
    const int wid   = threadIdx.x >> 5;
    const int lane  = threadIdx.x & 31;
    const int wtile = blockIdx.x * 4 + wid;     // 128 blocks x 4 waves = 512
    const int mT = wtile & 7;                   // 8 m-tiles (BATCH=128)
    const int nT = wtile >> 3;                  // 64 n-tiles (HID=1024)
    const int m0 = mT * 16, n0 = nT * 16;
    const int col = lane & 15;
    const int rowHi = (lane >> 4) * 8;

    v8f acc[4];
    #pragma unroll
    for (int g = 0; g < 4; ++g)
        #pragma unroll
        for (int r = 0; r < 8; ++r)
            acc[g][r] = __builtin_nontemporal_load(
                &Gin_t[(size_t)(m0 + rowHi + r) * G4H + g * HID + n0 + col]);

    for (int kt = 0; kt < KT32; ++kt) {
        v16bf a = load_a_frag(h_in + (size_t)m0 * HID + kt * 32, HID);
        #pragma unroll
        for (int g = 0; g < 4; ++g) {
            const int nTile = g * 64 + nT;      // tiles of the 4096-wide Whh
            v16bf b = load_b_frag(Whh_frag, nTile * KT32 + kt);
            acc[g] = wmma_bf16(a, b, acc[g]);
        }
    }

    #pragma unroll
    for (int r = 0; r < 8; ++r) {
        const int row = m0 + rowHi + r;
        const size_t idx = (size_t)row * HID + n0 + col;
        float iv = fast_sig(acc[0][r]);
        float fv = fast_sig(acc[1][r]);
        float gv = fast_tanh(acc[2][r]);
        float ov = fast_sig(acc[3][r]);
        float cn = fv * c[idx] + iv * gv;       // each tile owns its c elements
        c[idx]   = cn;
        float hn = ov * fast_tanh(cn);
        __builtin_nontemporal_store(hn, &out_t[idx]);
        h_out[idx] = (bf16_t)hn;
    }
}

// ---------- host ----------
extern "C" void kernel_launch(void* const* d_in, const int* in_sizes, int n_in,
                              void* d_out, int out_size, void* d_ws, size_t ws_size,
                              hipStream_t stream) {
    const int*   input_lines = (const int*)d_in[0];
    // d_in[1] target_lines unused by reference output
    const float* embed = (const float*)d_in[2];
    const float* W_ih  = (const float*)d_in[3];
    const float* W_hh  = (const float*)d_in[4];
    const float* b_ih  = (const float*)d_in[5];
    const float* b_hh  = (const float*)d_in[6];
    float* out = (float*)d_out;

    // workspace layout (all region sizes are multiples of 256 B)
    char* p = (char*)d_ws;
    float*  Gin      = (float*)p;  p += (size_t)SEQ * BATCH * G4H * sizeof(float);   // 536.9 MB
    bf16_t* Wih_frag = (bf16_t*)p; p += (size_t)G4H * HID * sizeof(bf16_t);          //   8.4 MB
    bf16_t* Whh_frag = (bf16_t*)p; p += (size_t)G4H * HID * sizeof(bf16_t);          //   8.4 MB
    bf16_t* Xbf      = (bf16_t*)p; p += (size_t)SEQ * BATCH * HID * sizeof(bf16_t);  //  67.1 MB
    bf16_t* hbuf0    = (bf16_t*)p; p += (size_t)BATCH * HID * sizeof(bf16_t);
    bf16_t* hbuf1    = (bf16_t*)p; p += (size_t)BATCH * HID * sizeof(bf16_t);
    float*  cbuf     = (float*)p;  p += (size_t)BATCH * HID * sizeof(float);
    float*  bias     = (float*)p;  p += (size_t)G4H * sizeof(float);

    // 1) zero h0/c, fold biases (131072 threads exact)
    init_state_kernel<<<512, 256, 0, stream>>>(cbuf, hbuf0, bias, b_ih, b_hh);

    // 2) weights -> bf16 WMMA-B fragments (4,194,304 elements each)
    convert_weight_kernel<<<16384, 256, 0, stream>>>(W_ih, Wih_frag);
    convert_weight_kernel<<<16384, 256, 0, stream>>>(W_hh, Whh_frag);

    // 3) embedding gather -> bf16 X (33,554,432 elements, 4/thread)
    gather_embed_kernel<<<32768, 256, 0, stream>>>(input_lines, embed, Xbf);

    // 4) parallel input projection: Gin = X @ Wih^T + bias (64x64 wave tiles)
    input_gemm_kernel<<<4096, 256, 0, stream>>>(Xbf, Wih_frag, bias, Gin);

    // 5) 256 recurrent steps (stream order = inter-step barrier; h ping-pong)
    for (int t = 0; t < SEQ; ++t) {
        const bf16_t* hin  = (t & 1) ? hbuf1 : hbuf0;
        bf16_t*       hout = (t & 1) ? hbuf0 : hbuf1;
        lstm_step_kernel<<<128, 128, 0, stream>>>(
            Gin + (size_t)t * BATCH * G4H, Whh_frag, hin, hout, cbuf,
            out + (size_t)t * BATCH * HID);
    }
}
